// PointPillarsNet_57071525429700
// MI455X (gfx1250) — matrix operand order repaired and checked by reading, not compiled, not run on hardware
//
#include <hip/hip_runtime.h>
#include <hip/hip_bf16.h>

// ---------------------------------------------------------------------------
// PointPillars forward for gfx1250 (MI455X).
// Convs / deconvs / head = implicit-GEMM with v_wmma_f32_16x16x32_bf16.
// bf16 activations end-to-end; weights pre-swizzled into WMMA B-fragment
// order so B-tiles are linear 4KB blocks (loaded via TDM tensor_load_to_lds
// when available, vectorized copy otherwise).
// ---------------------------------------------------------------------------

typedef __attribute__((ext_vector_type(16))) __bf16 v16bf;
typedef __attribute__((ext_vector_type(8)))  float  v8f;
typedef __attribute__((ext_vector_type(4)))  unsigned int v4u;
typedef __attribute__((ext_vector_type(8)))  int v8i;
typedef __attribute__((ext_vector_type(4)))  int v4i;

#define NB 2
#define NP 12000
#define GW 432
#define GH 496

#if defined(__gfx1250__) && __has_builtin(__builtin_amdgcn_tensor_load_to_lds)
#define CDNA5_TDM 1
#else
#define CDNA5_TDM 0
#endif

// ---------------------------------------------------------------------------
// Utility kernels
// ---------------------------------------------------------------------------
__global__ void fill_zero_kernel(float* __restrict__ p, long n) {
  long i = (long)blockIdx.x * blockDim.x + threadIdx.x;
  long stride = (long)gridDim.x * blockDim.x;
  for (; i < n; i += stride) p[i] = 0.0f;
}

__global__ void cvt_bf16_kernel(const float* __restrict__ src, __bf16* __restrict__ dst, long n) {
  long i = (long)blockIdx.x * blockDim.x + threadIdx.x;
  long stride = (long)gridDim.x * blockDim.x;
  for (; i < n; i += stride) dst[i] = (__bf16)src[i];
}

// Pack conv weights into WMMA-B-fragment-ordered 4KB blocks:
// dst[((tap*kcN + kc)*nTiles + coTile)*2048 + swz(k,n)]
// swz: lane=(n%16)|((k/16)<<4), elem=k%16, group=n/16 -> (group*32+lane)*16+elem
// iohw=0: w is OIHW [Cout][Cin][K2];  iohw=1: w is IOHW [Cin][Cout][K2]
__global__ void pack_conv_kernel(const float* __restrict__ w, __bf16* __restrict__ dst,
                                 int Cin, int Cout, int K2, int nTiles, int iohw, int total) {
  int i = blockIdx.x * blockDim.x + threadIdx.x;
  if (i >= total) return;
  int r = i & 2047;
  int blk = i >> 11;
  int coTile = blk % nTiles;
  int t2 = blk / nTiles;
  int kcN = Cin >> 5;
  int kc = t2 % kcN;
  int tap = t2 / kcN;
  int g = r >> 9;
  int lane = (r >> 4) & 31;
  int e = r & 15;
  int n = g * 16 + (lane & 15);
  int k = ((lane >> 4) << 4) + e;
  int ci = kc * 32 + k;
  int co = coTile * 64 + n;
  float v = 0.0f;
  if (co < Cout)
    v = iohw ? w[((size_t)ci * Cout + co) * K2 + tap]
             : w[((size_t)co * Cin + ci) * K2 + tap];
  dst[i] = (__bf16)v;
}

// head: concat cls(18)/reg(42)/dir(12) 1x1 weights -> fragment blocks + bias[72]
__global__ void pack_head_kernel(const float* __restrict__ cw, const float* __restrict__ cb,
                                 const float* __restrict__ rw, const float* __restrict__ rb,
                                 const float* __restrict__ dw, const float* __restrict__ db,
                                 __bf16* __restrict__ dst, float* __restrict__ bias) {
  int i = blockIdx.x * blockDim.x + threadIdx.x;
  if (i < 12 * 2 * 2048) {
    int r = i & 2047;
    int blk = i >> 11;
    int coTile = blk & 1;
    int kc = blk >> 1;
    int g = r >> 9;
    int lane = (r >> 4) & 31;
    int e = r & 15;
    int n = g * 16 + (lane & 15);
    int k = ((lane >> 4) << 4) + e;
    int ci = kc * 32 + k;
    int co = coTile * 64 + n;
    float v = 0.0f;
    if (co < 72) {
      if (co < 18)      v = cw[(size_t)co * 384 + ci];
      else if (co < 60) v = rw[(size_t)(co - 18) * 384 + ci];
      else              v = dw[(size_t)(co - 60) * 384 + ci];
    }
    dst[i] = (__bf16)v;
  }
  if (i < 72) bias[i] = (i < 18) ? cb[i] : (i < 60) ? rb[i - 18] : db[i - 60];
}

// ---------------------------------------------------------------------------
// Pillar Feature Net
// ---------------------------------------------------------------------------
__global__ __launch_bounds__(64)
void pfn_kernel(const float* __restrict__ pillars, const int* __restrict__ npoints,
                const int* __restrict__ cx, const int* __restrict__ cy,
                const float* __restrict__ enc_w, const float* __restrict__ enc_scale,
                const float* __restrict__ enc_bias, float* __restrict__ pfeat) {
  __shared__ float feat[32][9];
  const int pid = blockIdx.x;
  const int t = threadIdx.x;
  const int np = npoints[pid];
  if (t < 32) {
    const float* pt = pillars + ((size_t)pid * 32 + t) * 4;
    float x = pt[0], y = pt[1], z = pt[2], r = pt[3];
    float m = (t < np) ? 1.0f : 0.0f;
    float sx = x * m, sy = y * m, sz = z * m;
#pragma unroll
    for (int off = 16; off > 0; off >>= 1) {
      sx += __shfl_xor(sx, off, 32);
      sy += __shfl_xor(sy, off, 32);
      sz += __shfl_xor(sz, off, 32);
    }
    float cnt = (float)(np < 1 ? 1 : np);
    float mx = sx / cnt, my = sy / cnt, mz = sz / cnt;
    float xc = (float)cx[pid] * 0.16f + 0.08f + 0.0f;
    float yc = (float)cy[pid] * 0.16f + 0.08f - 39.68f;
    feat[t][0] = x * m;        feat[t][1] = y * m;        feat[t][2] = z * m;
    feat[t][3] = r * m;        feat[t][4] = (x - mx) * m; feat[t][5] = (y - my) * m;
    feat[t][6] = (z - mz) * m; feat[t][7] = (x - xc) * m; feat[t][8] = (y - yc) * m;
  }
  __syncthreads();
  float w[9];
#pragma unroll
  for (int k = 0; k < 9; ++k) w[k] = enc_w[k * 64 + t];
  const float sc = enc_scale[t], bi = enc_bias[t];
  float best = 0.0f;  // ReLU outputs are >= 0
#pragma unroll 4
  for (int n = 0; n < 32; ++n) {
    float z = 0.0f;
#pragma unroll
    for (int k = 0; k < 9; ++k) z += feat[n][k] * w[k];
    z = fmaxf(z * sc + bi, 0.0f);
    best = fmaxf(best, z);
  }
  pfeat[(size_t)pid * 64 + t] = best;
}

// scatter-max (values >= 0, canvas zeroed -> int-bit atomicMax is exact)
__global__ void scatter_kernel(const float* __restrict__ pfeat, const int* __restrict__ cx,
                               const int* __restrict__ cy, float* __restrict__ canvas) {
  int i = blockIdx.x * blockDim.x + threadIdx.x;
  if (i >= NB * NP * 64) return;
  int pid = i >> 6;
  int c = i & 63;
  int b = pid / NP;
  int x = cx[pid], y = cy[pid];
  float v = pfeat[i];
  size_t off = (((size_t)b * 64 + c) * GH + y) * GW + x;
  atomicMax(reinterpret_cast<int*>(canvas) + off, __float_as_int(v));
}

// ---------------------------------------------------------------------------
// B-tile staging: one linear 4KB fragment-ordered block -> LDS
// ---------------------------------------------------------------------------
__device__ __forceinline__ void stage_b_tile(const __bf16* __restrict__ blk,
                                             __bf16* __restrict__ Bbuf, int t) {
#if CDNA5_TDM
  if (t < 32) {  // wave 0 drives the Tensor Data Mover
    unsigned long long ga = (unsigned long long)(size_t)blk;
    unsigned ldsa = (unsigned)(size_t)(void*)&Bbuf[0];
    // D# group0: count=1, lds_addr, global_addr[56:0], type=2
    v4u g0 = {1u, ldsa, (unsigned)ga, (unsigned)((ga >> 32) | (2u << 30))};
    // D# group1: data_size=2B; tensor_dim0=2048, tensor_dim1=1;
    // tile_dim0=2048, tile_dim1=1; tensor_dim0_stride=2048
    v8i g1 = {(int)(1u << 16), (int)(2048u << 16), (int)(1u << 16),
              (int)(2048u << 16), 1, 2048, 0, 0};
    v4i gz4 = {0, 0, 0, 0};
    v8i gz8 = {0, 0, 0, 0, 0, 0, 0, 0};
    __builtin_amdgcn_tensor_load_to_lds(g0, g1, gz4, gz4, gz8, 0);
    __builtin_amdgcn_s_wait_tensorcnt(0);
  }
#else
  *(int4*)&Bbuf[t * 8] = *(const int4*)(blk + (size_t)t * 8);
#endif
}

// ---------------------------------------------------------------------------
// Implicit-GEMM conv (3x3 or 1x1), WMMA bf16. Block tile: 128 px x 64 ch.
// Each thread owns one pixel + one k-octet pair = one A fragment row.
// ---------------------------------------------------------------------------
template <bool BF16OUT>
__global__ __launch_bounds__(256)
void conv_wmma_kernel(const __bf16* __restrict__ in, const __bf16* __restrict__ wt,
                      const float* __restrict__ scale, const float* __restrict__ bias,
                      void* __restrict__ outv,
                      int Cin, int Hin, int Win, int Hout, int Wout,
                      int stride, int ksize, int pad, int coutTotal, int bnrelu) {
  __shared__ __align__(32) __bf16 Abuf[4096];  // 128x32, fragment-swizzled
  __shared__ __align__(16) __bf16 Bbuf[2048];  // 32x64, fragment-swizzled
  const int t = threadIdx.x;
  const int M = Hout * Wout;
  const int HWin = Hin * Win;
  in += (size_t)blockIdx.z * Cin * HWin;
  const int m0 = blockIdx.x * 128;

  const int m = t & 127;
  const int khi = t >> 7;        // 0/1: this thread fills k {0-15} / {16-31}
  const int kh = khi << 4;
  const int p = m0 + m;
  const bool pv = (p < M);
  const int pp = pv ? p : 0;
  const int yo = pp / Wout, xo = pp - yo * Wout;
  const int ybase = yo * stride - pad;
  const int xbase = xo * stride - pad;
  const int aoff = (((m >> 4) * 32) + (m & 15) + (khi << 4)) * 16;

  const int lane = t & 31;
  const int wv = t >> 5;
  const int mg = (wv & 3) * 32;
  const int ng = (wv >> 2) * 32;

  v8f c00 = {}, c01 = {}, c10 = {}, c11 = {};
  const int K2 = ksize * ksize;
  const int kcN = Cin >> 5;

  for (int tap = 0; tap < K2; ++tap) {
    const int ty = tap / ksize;
    const int tx = tap % ksize;
    const int yi = ybase + ty;
    const int xi = xbase + tx;
    const bool ok = pv && yi >= 0 && yi < Hin && xi >= 0 && xi < Win;
    for (int kc = 0; kc < kcN; ++kc) {
      // ---- A tile: one coalesced channel-strided fragment row per thread
      const __bf16* src = in + ((size_t)(kc * 32 + kh)) * HWin + yi * Win + xi;
      v16bf av;
#pragma unroll
      for (int e = 0; e < 8; ++e) {
        av[e]     = ok ? src[(size_t)e * HWin] : (__bf16)0.0f;
        av[e + 8] = ok ? src[(size_t)(8 + e) * HWin] : (__bf16)0.0f;
      }
      *(v16bf*)&Abuf[aoff] = av;
      // ---- B tile: linear pre-swizzled 4KB block
      const __bf16* blk = wt + ((size_t)((tap * kcN + kc) * gridDim.y + blockIdx.y) << 11);
      stage_b_tile(blk, Bbuf, t);
      if (kc + 1 < kcN) __builtin_prefetch(blk + ((size_t)gridDim.y << 11), 0, 0);
      __syncthreads();

      v16bf a0 = *(const v16bf*)&Abuf[(((mg) >> 4) * 32 + lane) * 16];
      v16bf a1 = *(const v16bf*)&Abuf[(((mg + 16) >> 4) * 32 + lane) * 16];
      v16bf b0 = *(const v16bf*)&Bbuf[(((ng) >> 4) * 32 + lane) * 16];
      v16bf b1 = *(const v16bf*)&Bbuf[(((ng + 16) >> 4) * 32 + lane) * 16];
      c00 = __builtin_amdgcn_wmma_f32_16x16x32_bf16(false, a0, false, b0, (short)0, c00, false, false);
      c01 = __builtin_amdgcn_wmma_f32_16x16x32_bf16(false, a0, false, b1, (short)0, c01, false, false);
      c10 = __builtin_amdgcn_wmma_f32_16x16x32_bf16(false, a1, false, b0, (short)0, c10, false, false);
      c11 = __builtin_amdgcn_wmma_f32_16x16x32_bf16(false, a1, false, b1, (short)0, c11, false, false);
      __syncthreads();
    }
  }

  // ---- epilogue: C/D layout: elem r -> M=r+8*(lane>=16), N=lane%16
  const int ncol = lane & 15;
  const int mbase = (lane >> 4) * 8;
  v8f accs[4] = {c00, c01, c10, c11};
#pragma unroll
  for (int q = 0; q < 4; ++q) {
    int ms = (q >> 1) * 16;
    int ns = (q & 1) * 16;
    int co = blockIdx.y * 64 + ng + ns + ncol;
    if (co >= coutTotal) continue;
    float sc = bnrelu ? scale[co] : 1.0f;
    float bi = bias[co];
#pragma unroll
    for (int r = 0; r < 8; ++r) {
      int pq = m0 + mg + ms + mbase + r;
      if (pq < M) {
        float v = accs[q][r];
        v = bnrelu ? fmaxf(v * sc + bi, 0.0f) : (v + bi);
        size_t oi = (size_t)blockIdx.z * coutTotal * M + (size_t)co * M + pq;
        if (BF16OUT) ((__bf16*)outv)[oi] = (__bf16)v;
        else         ((float*)outv)[oi] = v;
      }
    }
  }
}

// ---------------------------------------------------------------------------
// Transposed conv (kernel==stride): per-tap 1x1 GEMM + strided scatter to f2d
// ---------------------------------------------------------------------------
__global__ __launch_bounds__(256)
void deconv_wmma_kernel(const __bf16* __restrict__ in, const __bf16* __restrict__ wt,
                        const float* __restrict__ scale, const float* __restrict__ bias,
                        __bf16* __restrict__ f2d,
                        int Cin, int Hin, int Win, int kk, int chOff) {
  __shared__ __align__(32) __bf16 Abuf[4096];
  __shared__ __align__(16) __bf16 Bbuf[2048];
  const int K2 = kk * kk;
  const int z = blockIdx.z;
  const int b = z / K2;
  const int tap = z % K2;
  const int dy = tap / kk, dx = tap % kk;
  const int t = threadIdx.x;
  const int M = Hin * Win;
  const int Ho = Hin * kk, Wo = Win * kk;
  in += (size_t)b * Cin * M;
  __bf16* out = f2d + ((size_t)b * 384 + chOff) * (size_t)Ho * Wo;
  const int m0 = blockIdx.x * 128;

  const int m = t & 127;
  const int khi = t >> 7;
  const int kh = khi << 4;
  const int p = m0 + m;
  const bool pv = (p < M);
  const int aoff = (((m >> 4) * 32) + (m & 15) + (khi << 4)) * 16;

  const int lane = t & 31;
  const int wv = t >> 5;
  const int mg = (wv & 3) * 32;
  const int ng = (wv >> 2) * 32;

  v8f c00 = {}, c01 = {}, c10 = {}, c11 = {};
  const int kcN = Cin >> 5;

  for (int kc = 0; kc < kcN; ++kc) {
    const __bf16* src = in + ((size_t)(kc * 32 + kh)) * M + p;
    v16bf av;
#pragma unroll
    for (int e = 0; e < 8; ++e) {
      av[e]     = pv ? src[(size_t)e * M] : (__bf16)0.0f;
      av[e + 8] = pv ? src[(size_t)(8 + e) * M] : (__bf16)0.0f;
    }
    *(v16bf*)&Abuf[aoff] = av;
    const __bf16* blk = wt + ((size_t)((tap * kcN + kc) * gridDim.y + blockIdx.y) << 11);
    stage_b_tile(blk, Bbuf, t);
    if (kc + 1 < kcN) __builtin_prefetch(blk + ((size_t)gridDim.y << 11), 0, 0);
    __syncthreads();

    v16bf a0 = *(const v16bf*)&Abuf[(((mg) >> 4) * 32 + lane) * 16];
    v16bf a1 = *(const v16bf*)&Abuf[(((mg + 16) >> 4) * 32 + lane) * 16];
    v16bf b0 = *(const v16bf*)&Bbuf[(((ng) >> 4) * 32 + lane) * 16];
    v16bf b1 = *(const v16bf*)&Bbuf[(((ng + 16) >> 4) * 32 + lane) * 16];
    c00 = __builtin_amdgcn_wmma_f32_16x16x32_bf16(false, a0, false, b0, (short)0, c00, false, false);
    c01 = __builtin_amdgcn_wmma_f32_16x16x32_bf16(false, a0, false, b1, (short)0, c01, false, false);
    c10 = __builtin_amdgcn_wmma_f32_16x16x32_bf16(false, a1, false, b0, (short)0, c10, false, false);
    c11 = __builtin_amdgcn_wmma_f32_16x16x32_bf16(false, a1, false, b1, (short)0, c11, false, false);
    __syncthreads();
  }

  const int ncol = lane & 15;
  const int mbase = (lane >> 4) * 8;
  v8f accs[4] = {c00, c01, c10, c11};
#pragma unroll
  for (int q = 0; q < 4; ++q) {
    int ms = (q >> 1) * 16;
    int ns = (q & 1) * 16;
    int co = blockIdx.y * 64 + ng + ns + ncol;
    float sc = scale[co];
    float bi = bias[co];
#pragma unroll
    for (int r = 0; r < 8; ++r) {
      int pq = m0 + mg + ms + mbase + r;
      if (pq < M) {
        int yi = pq / Win, xi = pq - yi * Win;
        int oy = yi * kk + dy, ox = xi * kk + dx;
        float v = fmaxf(accs[q][r] * sc + bi, 0.0f);
        out[(size_t)co * Ho * Wo + oy * Wo + ox] = (__bf16)v;
      }
    }
  }
}

// ---------------------------------------------------------------------------
// Host launcher
// ---------------------------------------------------------------------------
extern "C" void kernel_launch(void* const* d_in, const int* in_sizes, int n_in,
                              void* d_out, int out_size, void* d_ws, size_t ws_size,
                              hipStream_t stream) {
  (void)in_sizes; (void)n_in; (void)out_size; (void)ws_size;

  const float* pillars   = (const float*)d_in[0];
  const int*   npoints   = (const int*)d_in[1];
  const int*   coords_x  = (const int*)d_in[2];
  const int*   coords_y  = (const int*)d_in[3];
  const float* enc_w     = (const float*)d_in[4];
  const float* enc_scale = (const float*)d_in[5];
  const float* enc_bias  = (const float*)d_in[6];
  const float* bb_w[16];  for (int i = 0; i < 16; ++i) bb_w[i] = (const float*)d_in[7 + i];
  const float* bb_s[16];  for (int i = 0; i < 16; ++i) bb_s[i] = (const float*)d_in[23 + i];
  const float* bb_b[16];  for (int i = 0; i < 16; ++i) bb_b[i] = (const float*)d_in[39 + i];
  const float* nk_w[3];   for (int i = 0; i < 3; ++i) nk_w[i] = (const float*)d_in[55 + i];
  const float* nk_s[3];   for (int i = 0; i < 3; ++i) nk_s[i] = (const float*)d_in[58 + i];
  const float* nk_b[3];   for (int i = 0; i < 3; ++i) nk_b[i] = (const float*)d_in[61 + i];
  const float* hcw = (const float*)d_in[64];
  const float* hcb = (const float*)d_in[65];
  const float* hrw = (const float*)d_in[66];
  const float* hrb = (const float*)d_in[67];
  const float* hdw = (const float*)d_in[68];
  const float* hdb = (const float*)d_in[69];
  float* outp = (float*)d_out;

  // -------- workspace: f32 region then bf16 region
  float* wsf = (float*)d_ws;
  size_t o = 0;
  float* pfeat   = wsf + o; o += (size_t)NB * NP * 64;
  float* canvasF = wsf + o; o += (size_t)NB * 64 * GH * GW;
  float* headb   = wsf + o; o += 72;
  __bf16* hb = (__bf16*)(wsf + o);
  size_t ho = 0;
  __bf16* canvasH = hb + ho; ho += (size_t)NB * 64 * GH * GW;   // 27.4M
  __bf16* T0v = hb + ho; ho += 6856704;
  __bf16* T1v = hb + ho; ho += 6856704;
  __bf16* O1v = hb + ho; ho += 6856704;
  __bf16* O2v = hb + ho; ho += 3428352;
  __bf16* O3v = hb + ho; ho += 1714176;
  __bf16* f2d = hb + ho; ho += (size_t)NB * 384 * 248 * 216;    // 41.1M
  __bf16* bfw = hb + ho;

  // -------- backbone conv configs
  static const int cins[16]  = {64,64,64,64, 64,128,128,128,128,128, 128,256,256,256,256,256};
  static const int couts[16] = {64,64,64,64, 128,128,128,128,128,128, 256,256,256,256,256,256};
  static const int strd[16]  = {2,1,1,1, 2,1,1,1,1,1, 2,1,1,1,1,1};
  static const int hins[16]  = {496,248,248,248, 248,124,124,124,124,124, 124,62,62,62,62,62};
  static const int wins[16]  = {432,216,216,216, 216,108,108,108,108,108, 108,54,54,54,54,54};
  static const int houts[16] = {248,248,248,248, 124,124,124,124,124,124, 62,62,62,62,62,62};
  static const int wouts[16] = {216,216,216,216, 108,108,108,108,108,108, 54,54,54,54,54,54};

  // -------- pack weights (fragment-ordered 4KB blocks)
  __bf16* bbwp[16];
  size_t bo = 0;
  for (int i = 0; i < 16; ++i) {
    bbwp[i] = bfw + bo;
    int nT = couts[i] / 64;
    int total = 9 * (cins[i] / 32) * nT * 2048;
    bo += total;
    pack_conv_kernel<<<(total + 255) / 256, 256, 0, stream>>>(bb_w[i], bbwp[i], cins[i], couts[i], 9, nT, 0, total);
  }
  static const int nkk[3] = {1, 2, 4};
  static const int nci[3] = {64, 128, 256};
  __bf16* nkwp[3];
  for (int i = 0; i < 3; ++i) {
    nkwp[i] = bfw + bo;
    int k2 = nkk[i] * nkk[i];
    int total = k2 * (nci[i] / 32) * 2 * 2048;
    bo += total;
    pack_conv_kernel<<<(total + 255) / 256, 256, 0, stream>>>(nk_w[i], nkwp[i], nci[i], 128, k2, 2, 1, total);
  }
  __bf16* headw = bfw + bo; bo += 12 * 2 * 2048;
  pack_head_kernel<<<(12 * 2 * 2048 + 255) / 256, 256, 0, stream>>>(hcw, hcb, hrw, hrb, hdw, hdb, headw, headb);

  // -------- PFN + scatter + bf16 canvas
  pfn_kernel<<<NB * NP, 64, 0, stream>>>(pillars, npoints, coords_x, coords_y,
                                         enc_w, enc_scale, enc_bias, pfeat);
  long canvas_n = (long)NB * 64 * GH * GW;
  fill_zero_kernel<<<4096, 256, 0, stream>>>(canvasF, canvas_n);
  scatter_kernel<<<(NB * NP * 64 + 255) / 256, 256, 0, stream>>>(pfeat, coords_x, coords_y, canvasF);
  cvt_bf16_kernel<<<4096, 256, 0, stream>>>(canvasF, canvasH, canvas_n);

  // -------- backbone (ping-pong T0/T1, stage outputs persist for neck)
  const __bf16* srcs[16] = {canvasH, T0v, T1v, T0v,
                            O1v, T0v, T1v, T0v, T1v, T0v,
                            O2v, T0v, T1v, T0v, T1v, T0v};
  __bf16* dsts[16] = {T0v, T1v, T0v, O1v,
                      T0v, T1v, T0v, T1v, T0v, O2v,
                      T0v, T1v, T0v, T1v, T0v, O3v};
  for (int i = 0; i < 16; ++i) {
    int M = houts[i] * wouts[i];
    dim3 grid((M + 127) / 128, couts[i] / 64, NB);
    conv_wmma_kernel<true><<<grid, 256, 0, stream>>>(srcs[i], bbwp[i], bb_s[i], bb_b[i], dsts[i],
                                                     cins[i], hins[i], wins[i], houts[i], wouts[i],
                                                     strd[i], 3, 1, couts[i], 1);
  }

  // -------- neck deconvs -> concat into f2d [NB][384][248][216] (bf16)
  const __bf16* nsrc[3] = {O1v, O2v, O3v};
  static const int nhin[3] = {248, 124, 62};
  static const int nwin[3] = {216, 108, 54};
  for (int i = 0; i < 3; ++i) {
    int M = nhin[i] * nwin[i];
    dim3 grid((M + 127) / 128, 2, NB * nkk[i] * nkk[i]);
    deconv_wmma_kernel<<<grid, 256, 0, stream>>>(nsrc[i], nkwp[i], nk_s[i], nk_b[i], f2d,
                                                 nci[i], nhin[i], nwin[i], nkk[i], i * 128);
  }

  // -------- head: 1x1 conv 384 -> 72 (bias only) -> d_out f32
  {
    int M = 248 * 216;
    dim3 grid((M + 127) / 128, 2, NB);
    conv_wmma_kernel<false><<<grid, 256, 0, stream>>>(f2d, headw, headb, headb, outp,
                                                      384, 248, 216, 248, 216,
                                                      1, 1, 0, 72, 0);
  }
}
